// InterActHead_66460323938664
// MI455X (gfx1250) — matrix operand order repaired and checked by reading, not compiled, not run on hardware
//
#include <hip/hip_runtime.h>
#include <hip/hip_bf16.h>

// ---------------------------------------------------------------------------
// Problem constants (from the reference)
// ---------------------------------------------------------------------------
#define Bc   16
#define L1c  768
#define L2c  768
#define AAc  1024
#define NCc  256
#define Hc   512
#define HHc  256
#define Lc   (L1c + L2c)          // 1536
#define Cqkv 1536                 // 6 * HH packed QKV columns

typedef __attribute__((ext_vector_type(16))) _Float16 v16h;
typedef __attribute__((ext_vector_type(8)))  _Float16 v8h;
typedef __attribute__((ext_vector_type(8)))  float    v8f;

// Tensor Data Mover availability (device pass only; fallback path otherwise)
#if defined(__HIP_DEVICE_COMPILE__) && __has_builtin(__builtin_amdgcn_tensor_load_to_lds) && __has_builtin(__builtin_amdgcn_s_wait_tensorcnt)
#define USE_TDM 1
typedef __attribute__((ext_vector_type(4))) unsigned int v4u;
typedef __attribute__((ext_vector_type(8))) int          v8i;
typedef __attribute__((ext_vector_type(4))) int          v4i;
#else
#define USE_TDM 0
#endif

// ---------------------------------------------------------------------------
// WMMA helpers (wave32, 16x16x32 f16 -> f32)
// ---------------------------------------------------------------------------
__device__ inline v8f v8f_zero() {
    v8f z;
#pragma unroll
    for (int i = 0; i < 8; ++i) z[i] = 0.0f;
    return z;
}

__device__ inline v8f wmma_f16(v16h a, v16h b, v8f c) {
    return __builtin_amdgcn_wmma_f32_16x16x32_f16(
        false, a, false, b, (short)0, c, false, false);
}

// A-matrix fragment, 16x32 (MxK), row-major source (row stride ld halves).
// Lane m=lane&15, kb=16*(lane>>4): halves 0..15 -> K=kb..kb+15.
__device__ inline v16h frag_a_ctg(const _Float16* __restrict__ base, int ld) {
    int lane = threadIdx.x & 31;
    int m    = lane & 15;
    int kb   = (lane >> 4) << 4;
    const _Float16* p = base + (size_t)m * ld + kb;
    v8h lo = *(const v8h*)(p);
    v8h hi = *(const v8h*)(p + 8);
    return __builtin_shufflevector(lo, hi, 0, 1, 2, 3, 4, 5, 6, 7,
                                           8, 9, 10, 11, 12, 13, 14, 15);
}

// B-matrix fragment, 32x16 (KxN), from a TRANSPOSED (N x K) view:
// element B[k][n] = base[n * ld + k]; 16 contiguous halves per lane.
__device__ inline v16h frag_b_ctg(const _Float16* __restrict__ base, int ld) {
    int lane = threadIdx.x & 31;
    int n    = lane & 15;
    int kb   = (lane >> 4) << 4;
    return *(const v16h*)(base + (size_t)n * ld + kb);
}

// ---------------------------------------------------------------------------
// Weight packing kernels: fp32 (K x N) -> f16 TRANSPOSED (N x K)
// ---------------------------------------------------------------------------
__global__ void transpose_to_f16_kernel(const float* __restrict__ src,
                                        _Float16* __restrict__ dst, int K, int N) {
    int i = blockIdx.x * blockDim.x + threadIdx.x;
    if (i >= K * N) return;
    int n = i / K;
    int k = i - n * K;
    dst[i] = (_Float16)src[k * N + n];
}

__global__ void pack_qkvT_kernel(const float* __restrict__ wq1, const float* __restrict__ wk1,
                                 const float* __restrict__ wv1, const float* __restrict__ wq2,
                                 const float* __restrict__ wk2, const float* __restrict__ wv2,
                                 _Float16* __restrict__ dst) {
    int i = blockIdx.x * blockDim.x + threadIdx.x;   // over 1536*512, dst-linear
    if (i >= Cqkv * Hc) return;
    int c   = i / Hc;
    int k   = i - c * Hc;
    int sel = c >> 8;
    int off = c & 255;
    const float* srcs[6] = { wq1, wk1, wv1, wq2, wk2, wv2 };
    dst[i] = (_Float16)srcs[sel][k * HHc + off];
}

__global__ void pack_bias_kernel(const float* __restrict__ bq1, const float* __restrict__ bk1,
                                 const float* __restrict__ bv1, const float* __restrict__ bq2,
                                 const float* __restrict__ bk2, const float* __restrict__ bv2,
                                 float* __restrict__ dst) {
    int i = blockIdx.x * blockDim.x + threadIdx.x;
    if (i >= Cqkv) return;
    const float* srcs[6] = { bq1, bk1, bv1, bq2, bk2, bv2 };
    dst[i] = srcs[i >> 8][i & 255];
}

// ---------------------------------------------------------------------------
// GEMM + LayerNorm:  out[M x 512] = LN(A[M x K] @ W) * g + b, Wt = (512 x K) f16.
// 16x128 A chunks staged f32->f16 in LDS (one barrier pair per 4 K-steps);
// 8 waves x 4 N-tiles.
// ---------------------------------------------------------------------------
__global__ __launch_bounds__(256)
void gemm_ln_kernel(const float* __restrict__ A, const _Float16* __restrict__ Wt,
                    const float* __restrict__ gamma, const float* __restrict__ beta,
                    _Float16* __restrict__ out, int K) {
    constexpr int N = Hc;  // 512
    __shared__ _Float16 sA[16 * 128];
    __shared__ float    sC[16 * N];
    __shared__ float    sRed[256];
    __shared__ float    sRed2[256];
    __shared__ float    sMean[16];
    __shared__ float    sRstd[16];

    int tid  = threadIdx.x;
    int lane = tid & 31;
    int wave = tid >> 5;
    size_t row0 = (size_t)blockIdx.x * 16;

    v8f acc[4];
#pragma unroll
    for (int t = 0; t < 4; ++t) acc[t] = v8f_zero();

    for (int kc0 = 0; kc0 < K; kc0 += 128) {
        __syncthreads();
        {   // Stage 16x128 chunk: 8 halves per thread (2 x float4 loads).
            int i  = tid << 3;                 // 0..2040
            int m  = i >> 7, kk = i & 127;
            const float* ap = A + (row0 + m) * (size_t)K + kc0 + kk;
            const float4 va = *(const float4*)(ap);
            const float4 vb = *(const float4*)(ap + 4);
            v8h h;
            h[0] = (_Float16)va.x; h[1] = (_Float16)va.y;
            h[2] = (_Float16)va.z; h[3] = (_Float16)va.w;
            h[4] = (_Float16)vb.x; h[5] = (_Float16)vb.y;
            h[6] = (_Float16)vb.z; h[7] = (_Float16)vb.w;
            *(v8h*)(sA + i) = h;
            if (kc0 + 128 < K)
                __builtin_prefetch(ap + 128, 0, 1);
        }
        __syncthreads();

#pragma unroll
        for (int ks = 0; ks < 4; ++ks) {
            int k0 = kc0 + ks * 32;
            v16h af = frag_a_ctg(sA + ks * 32, 128);
            v16h bf[4];
#pragma unroll
            for (int t = 0; t < 4; ++t) {
                int n0 = (wave * 4 + t) * 16;
                bf[t] = frag_b_ctg(Wt + (size_t)n0 * K + k0, K);
            }
#pragma unroll
            for (int t = 0; t < 4; ++t)
                acc[t] = wmma_f16(af, bf[t], acc[t]);
        }
    }

    // Spill the 16x512 tile into LDS for the LayerNorm epilogue.
    {
        int n  = lane & 15;
        int mb = (lane >> 4) << 3;
#pragma unroll
        for (int t = 0; t < 4; ++t) {
            int n0 = (wave * 4 + t) * 16;
#pragma unroll
            for (int r = 0; r < 8; ++r)
                sC[(mb + r) * N + n0 + n] = acc[t][r];
        }
    }
    __syncthreads();

    // Parallel LN stats: 16 threads per row, 32 elements each.
    {
        int m   = tid >> 4;
        int seg = tid & 15;
        float s = 0.0f, s2 = 0.0f;
        const float* rowp = sC + m * N + seg * 32;
#pragma unroll 8
        for (int c = 0; c < 32; ++c) { float v = rowp[c]; s += v; s2 += v * v; }
        sRed[tid] = s; sRed2[tid] = s2;
    }
    __syncthreads();
    if (tid < 16) {
        float s = 0.0f, s2 = 0.0f;
#pragma unroll
        for (int j = 0; j < 16; ++j) { s += sRed[tid * 16 + j]; s2 += sRed2[tid * 16 + j]; }
        float mu  = s * (1.0f / N);
        float var = s2 * (1.0f / N) - mu * mu;
        sMean[tid] = mu;
        sRstd[tid] = rsqrtf(var + 1e-5f);
    }
    __syncthreads();

    for (int i = tid; i < 16 * N; i += 256) {
        int m = i >> 9;
        int c = i & (N - 1);
        float v = (sC[i] - sMean[m]) * sRstd[m] * gamma[c] + beta[c];
        out[(row0 + m) * (size_t)N + c] = (_Float16)v;
    }
}

// ---------------------------------------------------------------------------
// emb gather: (B, L, H) f16 from h1/h2 per lens; 8 halves per thread.
// ---------------------------------------------------------------------------
__global__ void emb_kernel(const _Float16* __restrict__ h1, const _Float16* __restrict__ h2,
                           const int* __restrict__ lens1, const int* __restrict__ lens2,
                           _Float16* __restrict__ emb) {
    size_t v8idx = (size_t)blockIdx.x * blockDim.x + threadIdx.x;
    const size_t T8 = (size_t)Bc * Lc * Hc / 8;
    if (v8idx >= T8) return;
    size_t idx = v8idx * 8;
    int c = (int)(idx & (Hc - 1));
    size_t rl = idx >> 9;
    int j = (int)(rl % Lc);
    int b = (int)(rl / Lc);
    int l1 = lens1[b], l2 = lens2[b];
    v8h v;
#pragma unroll
    for (int i = 0; i < 8; ++i) v[i] = (_Float16)0.0f;
    if (j < l1) {
        int jj = j < (L1c - 1) ? j : (L1c - 1);
        v = *(const v8h*)(h1 + ((size_t)b * L1c + jj) * Hc + c);
    } else if (j < l1 + l2) {
        int jj = (j - l1) < (L2c - 1) ? (j - l1) : (L2c - 1);
        v = *(const v8h*)(h2 + ((size_t)b * L2c + jj) * Hc + c);
    }
    *(v8h*)(emb + idx) = v;
}

// ---------------------------------------------------------------------------
// QKV GEMM: [M x 1536] = emb[M x 512] @ Wqkv + bias; WqkvT is (1536 x 512).
// The 16-row A panel (16x512 f16 = 16 KB, contiguous in memory) is staged in
// LDS ONCE -- via the Tensor Data Mover when available -- so the K-loop runs
// with no barriers. V column tiles are written transposed to vT.
// ---------------------------------------------------------------------------
__global__ __launch_bounds__(256)
void gemm_qkv_kernel(const _Float16* __restrict__ A, const _Float16* __restrict__ WqkvT,
                     const float* __restrict__ bias, _Float16* __restrict__ qkv,
                     _Float16* __restrict__ vT) {
    constexpr int K = Hc, N = Cqkv;
    __shared__ _Float16 sA[16 * K];       // 16 KB

    int tid  = threadIdx.x;
    int lane = tid & 31;
    int wave = tid >> 5;
    size_t row0 = (size_t)blockIdx.x * 16;
    int b_idx = (int)(row0 / Lc);
    int l0    = (int)(row0 % Lc);

#if USE_TDM
    if (wave == 0) {
        // Tensor DMA descriptor (CDNA5 ISA):
        // 2D tile: tile_dim0=512 elems, tile_dim1=16 rows, data_size=2B,
        // tensor_dim0=512, tensor_dim0_stride=512, D# type=2.
        unsigned long long ga = (unsigned long long)(size_t)(A + row0 * (size_t)K);
        unsigned int la = (unsigned int)(size_t)
            (__attribute__((address_space(3))) char*)(void*)sA;
        v4u g0;
        g0[0] = 1u;                                        // count=1, user mode
        g0[1] = la;                                        // lds_addr
        g0[2] = (unsigned int)(ga & 0xffffffffull);        // global_addr[31:0]
        g0[3] = (unsigned int)((ga >> 32) & 0x01ffffffull) // global_addr[56:32]
              | 0x80000000u;                               // type=2 (bits 127:126)
        v8i g1;
        g1[0] = 0x00010000;            // data_size=1 (2 bytes)
        g1[1] = (int)(512u << 16);     // tensor_dim0[15:0] at bits 63:48
        g1[2] = (int)(16u << 16);      // tensor_dim1[15:0] at bits 95:80
        g1[3] = (int)(512u << 16);     // tile_dim0 at bits 127:112
        g1[4] = 16;                    // tile_dim1 (tile_dim2=0)
        g1[5] = 512;                   // tensor_dim0_stride[31:0]
        g1[6] = 0;
        g1[7] = 0;
        v4i zz; zz[0] = 0; zz[1] = 0; zz[2] = 0; zz[3] = 0;
#if __clang_major__ >= 23
        v8i z8; 
#pragma unroll
        for (int q = 0; q < 8; ++q) z8[q] = 0;
        __builtin_amdgcn_tensor_load_to_lds(g0, g1, zz, zz, z8, 0);
#else
        __builtin_amdgcn_tensor_load_to_lds(g0, g1, zz, zz, 0);
#endif
        __builtin_amdgcn_s_wait_tensorcnt(0);
    }
    __syncthreads();
#else
    {   // Cooperative panel stage: 8192 contiguous halves.
        const _Float16* src = A + row0 * (size_t)K;
#pragma unroll
        for (int i = tid * 8; i < 16 * K; i += 256 * 8)
            *(v8h*)(sA + i) = *(const v8h*)(src + i);
    }
    __syncthreads();
#endif

    v8f acc[12];
#pragma unroll
    for (int t = 0; t < 12; ++t) acc[t] = v8f_zero();

    for (int k0 = 0; k0 < K; k0 += 32) {
        v16h af = frag_a_ctg(sA + k0, K);
#pragma unroll
        for (int tt = 0; tt < 3; ++tt) {
            v16h bf[4];
#pragma unroll
            for (int j = 0; j < 4; ++j) {
                int n0 = (wave * 12 + tt * 4 + j) * 16;
                bf[j] = frag_b_ctg(WqkvT + (size_t)n0 * K + k0, K);
            }
#pragma unroll
            for (int j = 0; j < 4; ++j)
                acc[tt * 4 + j] = wmma_f16(af, bf[j], acc[tt * 4 + j]);
        }
    }

    int n  = lane & 15;
    int mb = (lane >> 4) << 3;
#pragma unroll
    for (int t = 0; t < 12; ++t) {
        int n0 = (wave * 12 + t) * 16;
        float bv = bias[n0 + n];
        bool isV = (n0 >= 512 && n0 < 768) || (n0 >= 1280);
        if (!isV) {
#pragma unroll
            for (int r = 0; r < 8; ++r)
                qkv[(row0 + mb + r) * (size_t)N + n0 + n] = (_Float16)(acc[t][r] + bv);
        } else {
            int at = (n0 >= 1280) ? 1 : 0;
            int d  = n0 + n - (at ? 1280 : 512);
            _Float16* vbase = vT + (((size_t)at * Bc + b_idx) * HHc + d) * Lc;
#pragma unroll
            for (int r = 0; r < 8; ++r)
                vbase[l0 + mb + r] = (_Float16)(acc[t][r] + bv);
        }
    }
}

// ---------------------------------------------------------------------------
// Flash-style attention. Grid: (96 q-tiles, B, 2 attn sets). 128 threads.
// ---------------------------------------------------------------------------
__global__ __launch_bounds__(128)
void attn_kernel(const _Float16* __restrict__ qkv, const _Float16* __restrict__ vT,
                 const int* __restrict__ lens1, const int* __restrict__ lens2,
                 float* __restrict__ ctx) {
    int qt   = blockIdx.x;
    int b    = blockIdx.y;
    int at   = blockIdx.z;
    int tid  = threadIdx.x;
    int lane = tid & 31;
    int wave = tid >> 5;

    int total = lens1[b] + lens2[b];
    if (total > Lc) total = Lc;
    int q0 = qt * 16;

    const _Float16* Qb = qkv + (size_t)b * Lc * Cqkv + at * 768;
    const _Float16* Kb = Qb + 256;
    const _Float16* Vt = vT + ((size_t)at * Bc + b) * (size_t)HHc * Lc;

    __shared__ float    sS[16 * 64];
    __shared__ _Float16 sP[16 * 64];
    __shared__ float    sM[16];
    __shared__ float    sL[16];
    __shared__ float    sF[16];

    v16h qf[8];
#pragma unroll
    for (int s = 0; s < 8; ++s)
        qf[s] = frag_a_ctg(Qb + (size_t)q0 * Cqkv + s * 32, Cqkv);

    v8f o[4];
#pragma unroll
    for (int t = 0; t < 4; ++t) o[t] = v8f_zero();

    if (tid < 16) { sM[tid] = -3.0e30f; sL[tid] = 0.0f; }
    __syncthreads();

    const float scale = 0.0625f;    // 1/sqrt(256)

    for (int kc = 0; kc < Lc; kc += 64) {
        int kcol = kc + wave * 16;

        // ---- S = (Q @ K^T) * scale + mask --------------------------------
        v8f s_acc = v8f_zero();
#pragma unroll
        for (int dh = 0; dh < 2; ++dh) {
            v16h kf[4];
#pragma unroll
            for (int j = 0; j < 4; ++j)
                kf[j] = frag_b_ctg(Kb + (size_t)kcol * Cqkv + (dh * 4 + j) * 32, Cqkv);
#pragma unroll
            for (int j = 0; j < 4; ++j)
                s_acc = wmma_f16(qf[dh * 4 + j], kf[j], s_acc);
        }
        {
            int n  = lane & 15;
            int mb = (lane >> 4) << 3;
#pragma unroll
            for (int r = 0; r < 8; ++r) {
                int key  = kcol + n;
                int qrow = q0 + mb + r;
                float sc = s_acc[r] * scale;
                float vq = (qrow < total) ? 1.0f : 0.0f;
                float vk = (key  < total) ? 1.0f : 0.0f;
                sc += (1.0f - vq * vk) * -10000.0f;
                sS[(mb + r) * 64 + wave * 16 + n] = sc;
            }
        }
        __syncthreads();

        // ---- online softmax bookkeeping (one thread per query row) ------
        if (tid < 16) {
            float mold = sM[tid];
            float mc   = mold;
#pragma unroll 4
            for (int c = 0; c < 64; ++c) mc = fmaxf(mc, sS[tid * 64 + c]);
            float f = __expf(mold - mc);
            float ssum = 0.0f;
#pragma unroll 4
            for (int c = 0; c < 64; ++c) {
                float p = __expf(sS[tid * 64 + c] - mc);
                sP[tid * 64 + c] = (_Float16)p;
                ssum += p;
            }
            sM[tid] = mc;
            sL[tid] = sL[tid] * f + ssum;
            sF[tid] = f;
        }
        __syncthreads();

        // ---- rescale running output, then O += P @ V ---------------------
        {
            int mb = (lane >> 4) << 3;
#pragma unroll
            for (int t = 0; t < 4; ++t)
#pragma unroll
                for (int r = 0; r < 8; ++r)
                    o[t][r] *= sF[mb + r];
        }
#pragma unroll
        for (int ks = 0; ks < 2; ++ks) {
            v16h pf = frag_a_ctg(sP + ks * 32, 64);
            v16h vf[4];
#pragma unroll
            for (int t = 0; t < 4; ++t) {
                int n0 = (wave * 4 + t) * 16;
                vf[t] = frag_b_ctg(Vt + (size_t)n0 * Lc + kc + ks * 32, Lc);
            }
#pragma unroll
            for (int t = 0; t < 4; ++t)
                o[t] = wmma_f16(pf, vf[t], o[t]);
        }
        __syncthreads();
    }

    // ---- epilogue: divide by row sum, write ctx --------------------------
    {
        int n  = lane & 15;
        int mb = (lane >> 4) << 3;
#pragma unroll
        for (int t = 0; t < 4; ++t) {
            int n0 = (wave * 4 + t) * 16;
#pragma unroll
            for (int r = 0; r < 8; ++r) {
                int qrow    = q0 + mb + r;
                float denom = sL[mb + r];
                float val   = (denom > 0.0f) ? (o[t][r] / denom) : 0.0f;
                ctx[((size_t)b * Lc + qrow) * (size_t)Hc + at * HHc + n0 + n] = val;
            }
        }
    }
}

// ---------------------------------------------------------------------------
// Head: pooled = mean_L(ctx); h = gelu(pooled @ W3); out = h @ Wp
// ---------------------------------------------------------------------------
__global__ __launch_bounds__(256)
void head_kernel(const float* __restrict__ ctx, const float* __restrict__ W3,
                 const float* __restrict__ Wp, float* __restrict__ out) {
    int b   = blockIdx.x;
    int tid = threadIdx.x;
    __shared__ float sPool[Hc];
    __shared__ float sH[64];

    for (int c = tid; c < Hc; c += 256) {
        float s = 0.0f;
        const float* p = ctx + (size_t)b * Lc * Hc + c;
#pragma unroll 4
        for (int l = 0; l < Lc; ++l) s += p[(size_t)l * Hc];
        sPool[c] = s * (1.0f / Lc);
    }
    __syncthreads();

    if (tid < 64) {
        float s = 0.0f;
#pragma unroll 4
        for (int k = 0; k < Hc; ++k) s += sPool[k] * W3[k * 64 + tid];
        sH[tid] = 0.5f * s * (1.0f + erff(s * 0.70710678f));
    }
    __syncthreads();

    if (tid == 0) {
        float s = 0.0f;
#pragma unroll 4
        for (int k = 0; k < 64; ++k) s += sH[k] * Wp[k];
        out[b] = s;
    }
}

// ---------------------------------------------------------------------------
// Launch
// ---------------------------------------------------------------------------
extern "C" void kernel_launch(void* const* d_in, const int* in_sizes, int n_in,
                              void* d_out, int out_size, void* d_ws, size_t ws_size,
                              hipStream_t stream) {
    const float* x1    = (const float*)d_in[0];
    const float* x2    = (const float*)d_in[1];
    const int*   lens1 = (const int*)d_in[2];
    const int*   lens2 = (const int*)d_in[3];
    const float* W1    = (const float*)d_in[4];
    const float* g1    = (const float*)d_in[5];
    const float* b1    = (const float*)d_in[6];
    const float* W2    = (const float*)d_in[7];
    const float* g2    = (const float*)d_in[8];
    const float* b2    = (const float*)d_in[9];
    const float* Wq1   = (const float*)d_in[10];
    const float* bq1   = (const float*)d_in[11];
    const float* Wk1   = (const float*)d_in[12];
    const float* bk1   = (const float*)d_in[13];
    const float* Wv1   = (const float*)d_in[14];
    const float* bv1   = (const float*)d_in[15];
    const float* Wq2   = (const float*)d_in[16];
    const float* bq2   = (const float*)d_in[17];
    const float* Wk2   = (const float*)d_in[18];
    const float* bk2   = (const float*)d_in[19];
    const float* Wv2   = (const float*)d_in[20];
    const float* bv2   = (const float*)d_in[21];
    const float* W3    = (const float*)d_in[22];
    const float* Wp    = (const float*)d_in[23];
    float* out = (float*)d_out;

    size_t off = 0;
    auto take = [&](size_t bytes) -> void* {
        void* p = (char*)d_ws + off;
        off += (bytes + 255) & ~(size_t)255;
        return p;
    };
    _Float16* w1t   = (_Float16*)take((size_t)AAc * Hc * 2);      // (512 x 1024)
    _Float16* w2t   = (_Float16*)take((size_t)NCc * Hc * 2);      // (512 x 256)
    _Float16* wqkvT = (_Float16*)take((size_t)Hc * Cqkv * 2);     // (1536 x 512)
    float*    bqkv  = (float*)take((size_t)Cqkv * 4);
    _Float16* h1    = (_Float16*)take((size_t)Bc * L1c * Hc * 2);
    _Float16* h2    = (_Float16*)take((size_t)Bc * L2c * Hc * 2);
    _Float16* emb   = (_Float16*)take((size_t)Bc * Lc * Hc * 2);
    _Float16* qkv   = (_Float16*)take((size_t)Bc * Lc * Cqkv * 2);
    _Float16* vT    = (_Float16*)take((size_t)2 * Bc * HHc * Lc * 2);
    float*    ctx   = (float*)take((size_t)Bc * Lc * Hc * 4);
    (void)ws_size; (void)n_in; (void)in_sizes; (void)out_size;

    // 1. Pack weights (transposed f16)
    transpose_to_f16_kernel<<<(AAc * Hc + 255) / 256, 256, 0, stream>>>(W1, w1t, AAc, Hc);
    transpose_to_f16_kernel<<<(NCc * Hc + 255) / 256, 256, 0, stream>>>(W2, w2t, NCc, Hc);
    pack_qkvT_kernel<<<(Hc * Cqkv + 255) / 256, 256, 0, stream>>>(Wq1, Wk1, Wv1, Wq2, Wk2, Wv2, wqkvT);
    pack_bias_kernel<<<(Cqkv + 255) / 256, 256, 0, stream>>>(bq1, bk1, bv1, bq2, bk2, bv2, bqkv);

    // 2. Encoder GEMM + LayerNorm
    gemm_ln_kernel<<<(Bc * L1c) / 16, 256, 0, stream>>>(x1, w1t, g1, b1, h1, AAc);
    gemm_ln_kernel<<<(Bc * L2c) / 16, 256, 0, stream>>>(x2, w2t, g2, b2, h2, NCc);

    // 3. Gather / concat
    {
        size_t tot8 = (size_t)Bc * Lc * Hc / 8;
        emb_kernel<<<(unsigned)((tot8 + 255) / 256), 256, 0, stream>>>(h1, h2, lens1, lens2, emb);
    }

    // 4. QKV projections (A panel via TDM when available; V written transposed)
    gemm_qkv_kernel<<<(Bc * Lc) / 16, 256, 0, stream>>>(emb, wqkvT, bqkv, qkv, vT);

    // 5. Attention (both sets)
    attn_kernel<<<dim3(Lc / 16, Bc, 2), 128, 0, stream>>>(qkv, vT, lens1, lens2, ctx);

    // 6. Pool + GELU + output
    head_kernel<<<Bc, 256, 0, stream>>>(ctx, W3, Wp, out);
}
